// Attention_7430293422356
// MI455X (gfx1250) — matrix operand order repaired and checked by reading, not compiled, not run on hardware
//
#include <hip/hip_runtime.h>
#include <stdint.h>

// Fused flash-attention + projection for gfx1250 (CDNA5, wave32).
// - Matrix math: V_WMMA_F32_16X16X32_F16 (f16 in, f32 accumulate)
// - K/V staging: Tensor Data Mover (tensor_load_to_lds), double-buffered,
//   synchronized with s_wait_tensorcnt
// - Softmax row reductions: DPP16 butterflies (no LDS / ds_bpermute)

typedef __attribute__((ext_vector_type(16))) _Float16 v16h;
typedef __attribute__((ext_vector_type(8)))  _Float16 v8h;
typedef __attribute__((ext_vector_type(4)))  _Float16 v4h;
typedef __attribute__((ext_vector_type(8)))  float    v8f;
typedef __attribute__((ext_vector_type(4)))  uint32_t v4u;
typedef __attribute__((ext_vector_type(8)))  int      v8i;
typedef __attribute__((ext_vector_type(4)))  int      v4i;

#define HEADS 8
#define DH    96     // head dim
#define SEQ   2048   // sequence length
#define CH    768    // channels
#define BATCH 4
#define TK    32     // keys per tile iteration
#define TM    32     // query rows per wave (2 m-blocks of 16)

__device__ __forceinline__ v8f wmma_f16(v16h a, v16h b, v8f c) {
  return __builtin_amdgcn_wmma_f32_16x16x32_f16(false, a, false, b, (short)0, c,
                                                false, false);
}

// ---- DPP16 16-lane butterfly reductions (VALU only, no DS traffic) --------
template <int CTRL>
__device__ __forceinline__ float dppf(float v) {
  return __builtin_bit_cast(
      float, __builtin_amdgcn_update_dpp(0, __builtin_bit_cast(int, v), CTRL,
                                         0xF, 0xF, true));
}
__device__ __forceinline__ float red_max16(float v) {
  v = fmaxf(v, dppf<0xB1>(v));    // quad_perm(1,0,3,2)
  v = fmaxf(v, dppf<0x4E>(v));    // quad_perm(2,3,0,1)
  v = fmaxf(v, dppf<0x141>(v));   // row_half_mirror
  v = fmaxf(v, dppf<0x140>(v));   // row_mirror
  return v;
}
__device__ __forceinline__ float red_sum16(float v) {
  v += dppf<0xB1>(v);
  v += dppf<0x4E>(v);
  v += dppf<0x141>(v);
  v += dppf<0x140>(v);
  return v;
}

// Issue one TDM descriptor: load a TK x DH f32 tile (row stride CH) from
// global into contiguous LDS at lds_off. D# layout per CDNA5 ISA §8.3/8.4.
__device__ __forceinline__ void tdm_load_tile(const float* gsrc,
                                              uint32_t lds_off) {
  uint64_t ga = (uint64_t)(uintptr_t)gsrc;
  v4u g0;
  g0[0] = 1u;                                    // count=1, user mode
  g0[1] = lds_off;                               // lds_addr (bytes)
  g0[2] = (uint32_t)ga;                          // global_addr[31:0]
  g0[3] = (uint32_t)((ga >> 32) & 0x01FFFFFFu)   // global_addr[56:32]
          | (2u << 30);                          // type = 2 ("image")
  v8i g1;
  g1[0] = (int)(2u << 16);                       // data_size = 2 -> 4 bytes
  g1[1] = (int)(((uint32_t)DH) << 16);           // tensor_dim0[15:0]
  g1[2] = (int)(((uint32_t)SEQ) << 16);          // dim0 hi16 | tensor_dim1 lo16
  g1[3] = (int)(((uint32_t)DH) << 16);           // dim1 hi16 | tile_dim0 = 96
  g1[4] = (int)TK;                               // tile_dim1 = 32, tile_dim2 = 0
  g1[5] = (int)CH;                               // tensor_dim0_stride = 768
  g1[6] = 0;
  g1[7] = 0;
  v4i zz = {0, 0, 0, 0};
#if defined(__clang_major__) && __clang_major__ >= 23
  v8i z8 = {0, 0, 0, 0, 0, 0, 0, 0};
  __builtin_amdgcn_tensor_load_to_lds(g0, g1, zz, zz, z8, 0);
#else
  __builtin_amdgcn_tensor_load_to_lds(g0, g1, zz, zz, 0);
#endif
}

// ---------------------------------------------------------------------------
// Kernel 1: fused attention. One wave handles 32 query rows of one (b,h).
// grid = B*H*(SEQ/32) = 2048 blocks, 32 threads each.
// ---------------------------------------------------------------------------
__global__ __launch_bounds__(32) void attn_fused_kernel(
    const float* __restrict__ x, float* __restrict__ att) {
  __shared__ alignas(16) float    stage[2][TK * DH];  // TDM f32 staging (2x12KB)
  __shared__ alignas(16) _Float16 qf[TM * DH];        // Q tile f16 (scale folded)
  __shared__ alignas(16) _Float16 kf[TK * DH];        // K tile, row-major 32x96
  __shared__ alignas(16) _Float16 vt[DH * TK];        // V tile, transposed 96x32
  __shared__ alignas(16) _Float16 pbuf[2][16 * TK];   // P tiles 16x32 per m-block

  const int lane = threadIdx.x;
  const int half = lane >> 4;
  const int l16  = lane & 15;

  const int tile = blockIdx.x & (SEQ / TM - 1);   // 0..63
  const int bh   = blockIdx.x / (SEQ / TM);
  const int b    = bh >> 3;
  const int h    = bh & (HEADS - 1);

  const float scale = rsqrtf((float)DH);
  const float* xbh  = x + (size_t)b * SEQ * CH + h * DH;
  const int row0    = tile * TM;

  // --- Q tile -> LDS as f16 with softmax scale folded in (cooperative)
#pragma unroll
  for (int it = 0; it < 24; ++it) {
    int f4 = it * 32 + lane;            // float4 index 0..767
    int r  = f4 / (DH / 4);             // row 0..31
    int c4 = (f4 - r * (DH / 4)) * 4;   // col 0..92
    float4 v = ((const float4*)(xbh + (size_t)(row0 + r) * CH))[c4 / 4];
    v4h h4;
    h4[0] = (_Float16)(v.x * scale);
    h4[1] = (_Float16)(v.y * scale);
    h4[2] = (_Float16)(v.z * scale);
    h4[3] = (_Float16)(v.w * scale);
    *(v4h*)&qf[r * DH + c4] = h4;
  }

  v8f o[2][DH / 16];
#pragma unroll
  for (int mb = 0; mb < 2; ++mb)
#pragma unroll
    for (int nb = 0; nb < DH / 16; ++nb) o[mb][nb] = (v8f){};

  float row_m[2][8], row_l[2][8];
#pragma unroll
  for (int mb = 0; mb < 2; ++mb)
#pragma unroll
    for (int r = 0; r < 8; ++r) {
      row_m[mb][r] = -3.0e38f;
      row_l[mb][r] = 0.0f;
    }

  const uint32_t stage_off0 = (uint32_t)(uintptr_t)&stage[0][0];
  const uint32_t stage_off1 = (uint32_t)(uintptr_t)&stage[1][0];

  tdm_load_tile(xbh, stage_off0);   // tile 0 in flight

  for (int jt = 0; jt < SEQ / TK; ++jt) {
    const int c0 = jt * TK;
    if (jt + 1 < SEQ / TK) {
      tdm_load_tile(xbh + (size_t)(c0 + TK) * CH,
                    ((jt + 1) & 1) ? stage_off1 : stage_off0);
    }
    __builtin_amdgcn_s_wait_tensorcnt(1);   // current tile landed in LDS
    __syncthreads();

    // --- convert f32 staging -> f16 K (row-major) and V^T (transposed)
    const float* sg = stage[jt & 1];
#pragma unroll
    for (int it = 0; it < 24; ++it) {
      int f4 = it * 32 + lane;
      int r  = f4 / (DH / 4);
      int c4 = (f4 - r * (DH / 4)) * 4;
      float4 v = ((const float4*)sg)[f4];
      _Float16 h0 = (_Float16)v.x, h1 = (_Float16)v.y;
      _Float16 h2 = (_Float16)v.z, h3 = (_Float16)v.w;
      v4h h4; h4[0] = h0; h4[1] = h1; h4[2] = h2; h4[3] = h3;
      *(v4h*)&kf[r * DH + c4] = h4;
      vt[(c4 + 0) * TK + r] = h0;
      vt[(c4 + 1) * TK + r] = h1;
      vt[(c4 + 2) * TK + r] = h2;
      vt[(c4 + 3) * TK + r] = h3;
    }
    __syncthreads();

#pragma unroll
    for (int mb = 0; mb < 2; ++mb) {
      // --- S = Q K^T : 2 key-blocks x 3 K-chunks = 6 WMMAs
      v8f s0 = (v8f){}, s1 = (v8f){};
#pragma unroll
      for (int c = 0; c < 3; ++c) {
        // A frag from LDS Q tile: elements 0..7 <- K c*32+half*16..,
        //                         elements 8..15 <- K c*32+half*16+8..
        v8h qlo = *(const v8h*)&qf[(mb * 16 + l16) * DH + c * 32 + half * 16];
        v8h qhi =
            *(const v8h*)&qf[(mb * 16 + l16) * DH + c * 32 + half * 16 + 8];
        v16h qa;
#pragma unroll
        for (int e = 0; e < 8; ++e) {
          qa[e]     = qlo[e];
          qa[8 + e] = qhi[e];
        }
        v16h bk0 = *(const v16h*)&kf[(0 + l16) * DH + c * 32 + half * 16];
        v16h bk1 = *(const v16h*)&kf[(16 + l16) * DH + c * 32 + half * 16];
        s0 = wmma_f16(qa, bk0, s0);
        s1 = wmma_f16(qa, bk1, s1);
      }

      // --- online softmax over 32 keys (DPP reductions, pure VALU)
#pragma unroll
      for (int r = 0; r < 8; ++r) {
        float sv0 = s0[r], sv1 = s1[r];
        float mx    = red_max16(fmaxf(sv0, sv1));
        float m_new = fmaxf(row_m[mb][r], mx);
        float alpha = __expf(row_m[mb][r] - m_new);
        float p0    = __expf(sv0 - m_new);
        float p1    = __expf(sv1 - m_new);
        float ps    = red_sum16(p0 + p1);
        row_l[mb][r] = row_l[mb][r] * alpha + ps;
        row_m[mb][r] = m_new;
#pragma unroll
        for (int nb = 0; nb < DH / 16; ++nb) o[mb][nb][r] *= alpha;
        pbuf[mb][(r + 8 * half) * TK + l16]      = (_Float16)p0;
        pbuf[mb][(r + 8 * half) * TK + 16 + l16] = (_Float16)p1;
      }
      __syncthreads();

      // --- P in A-layout (two b128 LDS loads)
      v8h plo = *(const v8h*)&pbuf[mb][l16 * TK + half * 16];
      v8h phi = *(const v8h*)&pbuf[mb][l16 * TK + half * 16 + 8];
      v16h pa;
#pragma unroll
      for (int e = 0; e < 8; ++e) {
        pa[e]     = plo[e];
        pa[8 + e] = phi[e];
      }

      // --- O += P V : 6 WMMAs (K=32 in one shot via transposed V tile)
#pragma unroll
      for (int nb = 0; nb < DH / 16; ++nb) {
        v16h bv = *(const v16h*)&vt[(nb * 16 + l16) * TK + half * 16];
        o[mb][nb] = wmma_f16(pa, bv, o[mb][nb]);
      }
    }
    __syncthreads();   // protect kf/vt before next tile's conversion
  }

  // --- normalize, scatter to att[b][n][h*96 + dd] (f32 workspace)
  float* obase = att + ((size_t)b * SEQ + row0) * CH + h * DH;
#pragma unroll
  for (int mb = 0; mb < 2; ++mb) {
#pragma unroll
    for (int r = 0; r < 8; ++r) {
      float inv = 1.0f / row_l[mb][r];
      int row   = mb * 16 + r + 8 * half;
#pragma unroll
      for (int nb = 0; nb < DH / 16; ++nb) {
        obase[(size_t)row * CH + nb * 16 + l16] = o[mb][nb][r] * inv;
      }
    }
  }
}

// ---------------------------------------------------------------------------
// Kernel 2: Y = Att(8192x768) @ W^T + bias, f16 WMMA / f32 accumulate.
// One wave per 16x64 output tile. grid = 512*12 = 6144 blocks.
// ---------------------------------------------------------------------------
__device__ __forceinline__ v16h cvt16(const float* p) {
  float4 q0 = ((const float4*)p)[0];
  float4 q1 = ((const float4*)p)[1];
  float4 q2 = ((const float4*)p)[2];
  float4 q3 = ((const float4*)p)[3];
  v16h a;
  a[0]  = (_Float16)q0.x; a[1]  = (_Float16)q0.y;
  a[2]  = (_Float16)q0.z; a[3]  = (_Float16)q0.w;
  a[4]  = (_Float16)q1.x; a[5]  = (_Float16)q1.y;
  a[6]  = (_Float16)q1.z; a[7]  = (_Float16)q1.w;
  a[8]  = (_Float16)q2.x; a[9]  = (_Float16)q2.y;
  a[10] = (_Float16)q2.z; a[11] = (_Float16)q2.w;
  a[12] = (_Float16)q3.x; a[13] = (_Float16)q3.y;
  a[14] = (_Float16)q3.z; a[15] = (_Float16)q3.w;
  return a;
}

__global__ __launch_bounds__(32) void proj_kernel(
    const float* __restrict__ att, const float* __restrict__ w,
    const float* __restrict__ bias, float* __restrict__ y) {
  const int lane = threadIdx.x;
  const int half = lane >> 4;
  const int l16  = lane & 15;

  const int ct   = blockIdx.x % (CH / 64);
  const int rt   = blockIdx.x / (CH / 64);
  const int row0 = rt * 16;
  const int col0 = ct * 64;

  v8f acc[4];
#pragma unroll
  for (int c = 0; c < 4; ++c) acc[c] = (v8f){};

  for (int kb = 0; kb < CH; kb += 32) {
    // A frag: att rows, elem e <- K kb+half*16+e (contiguous)
    const float* ap = att + (size_t)(row0 + l16) * CH + kb;
    float4 a0 = ((const float4*)(ap + half * 16))[0];
    float4 a1 = ((const float4*)(ap + half * 16))[1];
    float4 a2 = ((const float4*)(ap + half * 16))[2];
    float4 a3 = ((const float4*)(ap + half * 16))[3];
    v16h a;
    a[0]  = (_Float16)a0.x; a[1]  = (_Float16)a0.y;
    a[2]  = (_Float16)a0.z; a[3]  = (_Float16)a0.w;
    a[4]  = (_Float16)a1.x; a[5]  = (_Float16)a1.y;
    a[6]  = (_Float16)a1.z; a[7]  = (_Float16)a1.w;
    a[8]  = (_Float16)a2.x; a[9]  = (_Float16)a2.y;
    a[10] = (_Float16)a2.z; a[11] = (_Float16)a2.w;
    a[12] = (_Float16)a3.x; a[13] = (_Float16)a3.y;
    a[14] = (_Float16)a3.z; a[15] = (_Float16)a3.w;
#pragma unroll
    for (int c = 0; c < 4; ++c) {
      // B frag element e = W[col0+c*16+l16][kb + half*16 + e] (contiguous f32)
      const float* bp = w + (size_t)(col0 + c * 16 + l16) * CH + kb + half * 16;
      v16h bb = cvt16(bp);
      acc[c] = wmma_f16(a, bb, acc[c]);
    }
  }

#pragma unroll
  for (int c = 0; c < 4; ++c) {
    float bv = bias[col0 + c * 16 + l16];
#pragma unroll
    for (int r = 0; r < 8; ++r) {
      y[(size_t)(row0 + r + 8 * half) * CH + col0 + c * 16 + l16] =
          acc[c][r] + bv;
    }
  }
}

// ---------------------------------------------------------------------------
extern "C" void kernel_launch(void* const* d_in, const int* in_sizes, int n_in,
                              void* d_out, int out_size, void* d_ws,
                              size_t ws_size, hipStream_t stream) {
  const float* x  = (const float*)d_in[0];   // (4, 2048, 768) f32
  const float* pw = (const float*)d_in[1];   // (768, 768) f32
  const float* pb = (const float*)d_in[2];   // (768,) f32
  float* y   = (float*)d_out;                // (4, 2048, 768) f32
  float* att = (float*)d_ws;                 // 8192*768 f32 scratch (25.2 MB)

  (void)in_sizes; (void)n_in; (void)out_size; (void)ws_size;

  attn_fused_kernel<<<BATCH * HEADS * (SEQ / TM), 32, 0, stream>>>(x, att);
  proj_kernel<<<(BATCH * SEQ / 16) * (CH / 64), 32, 0, stream>>>(att, pw, pb, y);
}
